// BaselineNodeModel_9474697855367
// MI455X (gfx1250) — matrix operand order repaired and checked by reading, not compiled, not run on hardware
//
#include <hip/hip_runtime.h>
#include <hip/hip_bf16.h>

// ---------------------------------------------------------------------------
// CLRS MPNN on MI455X (gfx1250): bf16 WMMA for all GEMM-shaped work, fused
// per-edge message MLP + masked max so the [N,N,H] tensor never hits memory.
// N=768, H=128, L=2, OUT=40.
// ---------------------------------------------------------------------------

typedef __bf16 bf16_t;
typedef __attribute__((ext_vector_type(16))) __bf16 v16bf;
typedef __attribute__((ext_vector_type(8)))  __bf16 v8bf;
typedef __attribute__((ext_vector_type(8)))  float  v8f;

#define WMMA_BF16(a, b, c) \
    __builtin_amdgcn_wmma_f32_16x16x32_bf16(false, (a), false, (b), (short)0, (c), false, false)

// ---- fragment loaders -----------------------------------------------------
// A-matrix 16x32 bf16 (ISA 7.12.2): lanes 0-15 row=lane, K = k0+{0..7,16..23};
// lanes 16-31 row=lane-16, K = k0+{8..15,24..31}.
__device__ __forceinline__ v16bf load_afrag_global(const float* A, int lda, int row0,
                                                   int kc, int lane) {
    int r = row0 + (lane & 15);
    const float* p = A + r * lda + kc * 32 + ((lane & 16) ? 8 : 0);
    v16bf a;
#pragma unroll
    for (int e = 0; e < 8; ++e) a[e] = (__bf16)p[e];
#pragma unroll
    for (int e = 0; e < 8; ++e) a[8 + e] = (__bf16)p[16 + e];
    return a;
}

__device__ __forceinline__ v16bf load_afrag_lds(const bf16_t* tile, int rt, int kc, int lane) {
    int r = rt * 16 + (lane & 15);
    const bf16_t* p = tile + r * 128 + kc * 32 + ((lane & 16) ? 8 : 0);
    v8bf lo = *reinterpret_cast<const v8bf*>(p);
    v8bf hi = *reinterpret_cast<const v8bf*>(p + 16);
    return __builtin_shufflevector(lo, hi, 0, 1, 2, 3, 4, 5, 6, 7,
                                   8, 9, 10, 11, 12, 13, 14, 15);
}

// B-matrix 32x16 bf16: lane n = lane&15; lanes 0-15 hold K=k0+{0..7,16..23},
// lanes 16-31 hold K=k0+{8..15,24..31} (two 16B loads from WT[n][k] layout).
__device__ __forceinline__ v16bf load_bfrag(const bf16_t* WT, int K, int ntile,
                                            int kc, int lane) {
    int n  = ntile * 16 + (lane & 15);
    int k0 = kc * 32 + ((lane & 16) ? 8 : 0);
    v8bf lo = *reinterpret_cast<const v8bf*>(WT + n * K + k0);
    v8bf hi = *reinterpret_cast<const v8bf*>(WT + n * K + k0 + 16);
    return __builtin_shufflevector(lo, hi, 0, 1, 2, 3, 4, 5, 6, 7,
                                   8, 9, 10, 11, 12, 13, 14, 15);
}

// ---- weight prep: fp32 W[K][N] -> bf16 WT[N][K] ---------------------------
__global__ void wprep_kernel(const float* __restrict__ W, bf16_t* __restrict__ WT,
                             int K, int N) {
    int idx = blockIdx.x * 256 + threadIdx.x;
    if (idx < K * N) {
        int k = idx / N, n = idx % N;
        WT[n * K + k] = (__bf16)W[idx];
    }
}

__global__ void zfill_kernel(float* __restrict__ p, int rows, int cols, int ld) {
    int idx = blockIdx.x * 256 + threadIdx.x;
    if (idx < rows * cols) p[(idx / cols) * ld + (idx % cols)] = 0.f;
}

// ---- svec[k] = sum_h me_w[h][k]; cbase[k] = me_b[k] + mg_b[k] -------------
__global__ void cvec_kernel(const float* __restrict__ me_w, const float* __restrict__ me_b,
                            const float* __restrict__ mg_b,
                            float* __restrict__ svec, float* __restrict__ cbase) {
    int k = threadIdx.x;  // 128 threads
    float s = 0.f;
    for (int h = 0; h < 128; ++h) s += me_w[h * 128 + k];
    svec[k]  = s;
    cbase[k] = me_b[k] + mg_b[k];
}

// ---- general WMMA GEMM: out = act(A0@W0 [+ A1@W1] + bias0 [+ bias1]) ------
// M=768 (grid 48 row tiles), N=128 (4 waves x 2 n-tiles).
__global__ void __launch_bounds__(128) gemm_kernel(
    const float* __restrict__ A0, int lda0, int K0, const bf16_t* __restrict__ W0T,
    const float* __restrict__ bias0,
    const float* __restrict__ A1, int lda1, int K1, const bf16_t* __restrict__ W1T,
    const float* __restrict__ bias1,
    float* __restrict__ out, int ldo, int relu_act) {
    int lane = threadIdx.x & 31;
    int wave = threadIdx.x >> 5;  // 0..3
    int row0 = blockIdx.x * 16;
    int mhalf = (lane & 16) ? 8 : 0;

    for (int nt = wave; nt < 8; nt += 4) {
        v8f acc = {};
        for (int kc = 0; kc < (K0 >> 5); ++kc) {
            v16bf a = load_afrag_global(A0, lda0, row0, kc, lane);
            v16bf b = load_bfrag(W0T, K0, nt, kc, lane);
            acc = WMMA_BF16(a, b, acc);
        }
        if (A1 != nullptr) {
            for (int kc = 0; kc < (K1 >> 5); ++kc) {
                v16bf a = load_afrag_global(A1, lda1, row0, kc, lane);
                v16bf b = load_bfrag(W1T, K1, nt, kc, lane);
                acc = WMMA_BF16(a, b, acc);
            }
        }
        int n = nt * 16 + (lane & 15);
        float bias = bias0[n] + (bias1 ? bias1[n] : 0.f);
#pragma unroll
        for (int v = 0; v < 8; ++v) {
            float t = acc[v] + bias;
            if (relu_act) t = t > 0.f ? t : 0.f;
            out[(row0 + v + mhalf) * ldo + n] = t;
        }
    }
}

// ---- fused per-edge message MLP + masked max ------------------------------
// msgs[i,j] = relu(msg1[j] + msg2[i] + cbase + adj*svec)
// t1 = relu(msgs@W1 + b1); t2 = t1@W2 + b2; red[j] = max_{i: adj[i,j]} t2[i,j]
// One WG (8 waves) per 16-wide j tile; i in chunks of 8 (128 pair rows).
__global__ void __launch_bounds__(256) msgs_kernel(
    const float* __restrict__ msg1, const float* __restrict__ msg2,
    const int* __restrict__ adj,
    const float* __restrict__ cbase, const float* __restrict__ svec,
    const bf16_t* __restrict__ W1T, const float* __restrict__ b1,
    const bf16_t* __restrict__ W2T, const float* __restrict__ b2,
    float* __restrict__ red) {
    __shared__ bf16_t Atile[128 * 128];  // 32 KB: relu'd pre-messages (bf16)
    __shared__ bf16_t Ttile[128 * 128];  // 32 KB: relu'd MLP1 output (bf16)
    __shared__ int    sAdj[128];         // adjacency block: [i_local*16 + j_local]

    int tid  = threadIdx.x;
    int lane = tid & 31;
    int wv   = tid >> 5;  // wave = owned n-tile (feature block)
    int j0   = blockIdx.x * 16;
    int mhalf = (lane & 16) ? 8 : 0;

    // MLP weights live in VGPRs for the whole kernel (8 x v16bf = 64 VGPRs).
    v16bf B1[4], B2[4];
#pragma unroll
    for (int kc = 0; kc < 4; ++kc) {
        B1[kc] = load_bfrag(W1T, 128, wv, kc, lane);
        B2[kc] = load_bfrag(W2T, 128, wv, kc, lane);
    }
    float b1n = b1[wv * 16 + (lane & 15)];
    float b2n = b2[wv * 16 + (lane & 15)];

    // Build-phase invariants: thread owns one column c and one j-parity h0.
    // Cache msg1[j0+jl][c] + cbase[c] for its 8 jl values, and svec[c].
    int c  = tid & 127;
    int h0 = tid >> 7;  // 0 or 1 = parity of owned j_local
    float sv = svec[c];
    float cb = cbase[c];
    float m1v[8];
#pragma unroll
    for (int jj = 0; jj < 8; ++jj)
        m1v[jj] = msg1[(j0 + h0 + 2 * jj) * 128 + c] + cb;

    float rmax[8];
#pragma unroll
    for (int v = 0; v < 8; ++v) rmax[v] = -1.0e6f;  // CLRS -BIG_NUMBER

    for (int i0 = 0; i0 < 768; i0 += 8) {
        // Phase 0: stage the 8x16 adjacency block (128 distinct values).
        if (tid < 128) sAdj[tid] = adj[(i0 + (tid >> 4)) * 768 + j0 + (tid & 15)];
        __syncthreads();

        // Phase 1: build A tile. row r = i_local*16 + j_local.
        // Only global traffic: 8 streaming msg2 rows.
#pragma unroll
        for (int il = 0; il < 8; ++il) {
            float m2c = msg2[(i0 + il) * 128 + c];
#pragma unroll
            for (int jj = 0; jj < 8; ++jj) {
                int r = il * 16 + h0 + 2 * jj;
                float v = m1v[jj] + m2c;
                if (sAdj[r] != 0) v += sv;
                Atile[r * 128 + c] = (__bf16)(v > 0.f ? v : 0.f);
            }
        }
        __syncthreads();

        // Warm L0 for the next chunk while the WMMA phases run.
        if (i0 + 8 < 768 && tid < 8) {
            __builtin_prefetch(msg2 + (i0 + 8 + tid) * 128, 0, 1);
            __builtin_prefetch(adj + (i0 + 8 + tid) * 768 + j0, 0, 1);
        }

        // Phase 2: MLP1, wave wv computes its 16 output features for all rows.
#pragma unroll
        for (int rt = 0; rt < 8; ++rt) {
            v8f acc = {};
#pragma unroll
            for (int kc = 0; kc < 4; ++kc)
                acc = WMMA_BF16(load_afrag_lds(Atile, rt, kc, lane), B1[kc], acc);
#pragma unroll
            for (int v = 0; v < 8; ++v) {
                float t = acc[v] + b1n;
                t = t > 0.f ? t : 0.f;
                Ttile[(rt * 16 + v + mhalf) * 128 + wv * 16 + (lane & 15)] = (__bf16)t;
            }
        }
        __syncthreads();

        // Phase 3: MLP2 + masked max. Row-tile rt <-> i_local = rt, so the
        // reduction target (j,k) per (lane,vgpr) is i-invariant: pure reg max.
#pragma unroll
        for (int rt = 0; rt < 8; ++rt) {
            v8f acc = {};
#pragma unroll
            for (int kc = 0; kc < 4; ++kc)
                acc = WMMA_BF16(load_afrag_lds(Ttile, rt, kc, lane), B2[kc], acc);
#pragma unroll
            for (int v = 0; v < 8; ++v) {
                float val = acc[v] + b2n;
                if (sAdj[rt * 16 + v + mhalf] != 0) rmax[v] = fmaxf(rmax[v], val);
            }
        }
        __syncthreads();  // protects sAdj/Ttile before next chunk rewrites them
    }

    int k = wv * 16 + (lane & 15);
#pragma unroll
    for (int v = 0; v < 8; ++v) red[(j0 + v + mhalf) * 128 + k] = rmax[v];
}

// ---- layer norm over last dim (128), one block per row --------------------
__global__ void __launch_bounds__(128) ln_kernel(const float* __restrict__ x,
                                                 const float* __restrict__ g,
                                                 const float* __restrict__ b,
                                                 float* __restrict__ out, int ldo) {
    int row = blockIdx.x, t = threadIdx.x;
    __shared__ float s[128];
    float v = x[row * 128 + t];
    s[t] = v;
    __syncthreads();
    for (int off = 64; off > 0; off >>= 1) {
        if (t < off) s[t] += s[t + off];
        __syncthreads();
    }
    float mu = s[0] * (1.f / 128.f);
    __syncthreads();
    float d = v - mu;
    s[t] = d * d;
    __syncthreads();
    for (int off = 64; off > 0; off >>= 1) {
        if (t < off) s[t] += s[t + off];
        __syncthreads();
    }
    float var = s[0] * (1.f / 128.f);
    out[row * ldo + t] = d * rsqrtf(var + 1e-5f) * g[t] + b[t];
}

// ---- head: out = log_softmax(h@h2_w + h2_b), 128 -> 40 --------------------
__global__ void __launch_bounds__(128) head2_kernel(const float* __restrict__ hbuf,
                                                    const float* __restrict__ W,
                                                    const float* __restrict__ bias,
                                                    float* __restrict__ out) {
    int row = blockIdx.x, t = threadIdx.x;
    __shared__ float h[128];
    __shared__ float logits[40];
    __shared__ float lse;
    h[t] = hbuf[row * 128 + t];
    __syncthreads();
    if (t < 40) {
        float s = bias[t];
        for (int k = 0; k < 128; ++k) s += h[k] * W[k * 40 + t];
        logits[t] = s;
    }
    __syncthreads();
    if (t == 0) {
        float mx = logits[0];
        for (int c = 1; c < 40; ++c) mx = fmaxf(mx, logits[c]);
        float se = 0.f;
        for (int c = 0; c < 40; ++c) se += __expf(logits[c] - mx);
        lse = mx + __logf(se);
    }
    __syncthreads();
    if (t < 40) out[row * 40 + t] = logits[t] - lse;
}

// ---------------------------------------------------------------------------
extern "C" void kernel_launch(void* const* d_in, const int* in_sizes, int n_in,
                              void* d_out, int out_size, void* d_ws, size_t ws_size,
                              hipStream_t stream) {
    (void)in_sizes; (void)n_in; (void)out_size; (void)ws_size;
    const float* x    = (const float*)d_in[0];
    const int*   adj  = (const int*)d_in[1];
    const float* Wp   = (const float*)d_in[2];
    const float* bp   = (const float*)d_in[3];
    const float* m1_w = (const float*)d_in[4];
    const float* m1_b = (const float*)d_in[5];
    const float* m2_w = (const float*)d_in[6];
    const float* m2_b = (const float*)d_in[7];
    const float* me_w = (const float*)d_in[8];
    const float* me_b = (const float*)d_in[9];
    const float* mg_b = (const float*)d_in[11];  // mg_w dead: graph == 0
    const float* p1_w = (const float*)d_in[12];
    const float* p1_b = (const float*)d_in[13];
    const float* p2_w = (const float*)d_in[14];
    const float* p2_b = (const float*)d_in[15];
    const float* o1_w = (const float*)d_in[16];
    const float* o1_b = (const float*)d_in[17];
    const float* o2_w = (const float*)d_in[18];
    const float* o2_b = (const float*)d_in[19];
    const float* ln_g = (const float*)d_in[20];
    const float* ln_b = (const float*)d_in[21];
    const float* h1_w = (const float*)d_in[22];
    const float* h1_b = (const float*)d_in[23];
    const float* h2_w = (const float*)d_in[24];
    const float* h2_b = (const float*)d_in[25];

    char* ws = (char*)d_ws;
    float* z    = (float*)(ws + 0);         // [768,256] = [node | hidden]
    float* msg1 = (float*)(ws + 786432);    // [768,128]
    float* msg2 = (float*)(ws + 1179648);   // [768,128]
    float* red  = (float*)(ws + 1572864);   // [768,128]
    float* ret  = (float*)(ws + 1966080);   // [768,128]
    float* hbuf = (float*)(ws + 2359296);   // [768,128]
    float* svec = (float*)(ws + 2752512);   // [128]
    float* cbase = svec + 128;              // [128]

    bf16_t* WpT = (bf16_t*)(ws + 2753536);
    bf16_t* cur = WpT + 128 * 128;
    bf16_t *m1T[2], *m2T[2], *p1T[2], *p2T[2], *o1T[2], *o2T[2];
    for (int l = 0; l < 2; ++l) {
        m1T[l] = cur; cur += 256 * 128;
        m2T[l] = cur; cur += 256 * 128;
        p1T[l] = cur; cur += 128 * 128;
        p2T[l] = cur; cur += 128 * 128;
        o1T[l] = cur; cur += 256 * 128;
        o2T[l] = cur; cur += 128 * 128;
    }
    bf16_t* h1T = cur;

    auto wprep = [&](const float* W, bf16_t* WT, int K, int N) {
        wprep_kernel<<<(K * N + 255) / 256, 256, 0, stream>>>(W, WT, K, N);
    };
    wprep(Wp, WpT, 128, 128);
    for (int l = 0; l < 2; ++l) {
        wprep(m1_w + l * 256 * 128, m1T[l], 256, 128);
        wprep(m2_w + l * 256 * 128, m2T[l], 256, 128);
        wprep(p1_w + l * 128 * 128, p1T[l], 128, 128);
        wprep(p2_w + l * 128 * 128, p2T[l], 128, 128);
        wprep(o1_w + l * 256 * 128, o1T[l], 256, 128);
        wprep(o2_w + l * 128 * 128, o2T[l], 128, 128);
    }
    wprep(h1_w, h1T, 128, 128);

    // hidden(0) = 0 in the second half of z
    zfill_kernel<<<(768 * 128 + 255) / 256, 256, 0, stream>>>(z + 128, 768, 128, 256);

    // node = x @ Wp + bp  (no activation), written into z[:, :128]
    gemm_kernel<<<48, 128, 0, stream>>>(x, 128, 128, WpT, bp,
                                        nullptr, 0, 0, nullptr, nullptr,
                                        z, 256, 0);

    for (int l = 0; l < 2; ++l) {
        cvec_kernel<<<1, 128, 0, stream>>>(me_w + l * 128 * 128, me_b + l * 128,
                                           mg_b + l * 128, svec, cbase);
        gemm_kernel<<<48, 128, 0, stream>>>(z, 256, 256, m1T[l], m1_b + l * 128,
                                            nullptr, 0, 0, nullptr, nullptr,
                                            msg1, 128, 0);
        gemm_kernel<<<48, 128, 0, stream>>>(z, 256, 256, m2T[l], m2_b + l * 128,
                                            nullptr, 0, 0, nullptr, nullptr,
                                            msg2, 128, 0);
        msgs_kernel<<<48, 256, 0, stream>>>(msg1, msg2, adj, cbase, svec,
                                            p1T[l], p1_b + l * 128,
                                            p2T[l], p2_b + l * 128, red);
        // ret = relu(z@o1 + o1_b + red@o2 + o2_b)
        gemm_kernel<<<48, 128, 0, stream>>>(z, 256, 256, o1T[l], o1_b + l * 128,
                                            red, 128, 128, o2T[l], o2_b + l * 128,
                                            ret, 128, 1);
        // hidden = LN(ret) -> z[:, 128:]
        ln_kernel<<<768, 128, 0, stream>>>(ret, ln_g + l * 128, ln_b + l * 128,
                                           z + 128, 256);
    }

    // head: hbuf = relu(hidden @ h1_w + h1_b); out = log_softmax(hbuf@h2_w + h2_b)
    gemm_kernel<<<48, 128, 0, stream>>>(z + 128, 256, 128, h1T, h1_b,
                                        nullptr, 0, 0, nullptr, nullptr,
                                        hbuf, 128, 1);
    head2_kernel<<<768, 128, 0, stream>>>(hbuf, h2_w, h2_b, (float*)d_out);
}
